// GQAAttention_7095285973532
// MI455X (gfx1250) — matrix operand order repaired and checked by reading, not compiled, not run on hardware
//
#include <hip/hip_runtime.h>
#include <hip/hip_bf16.h>
#include <stdint.h>

// Problem constants (from the reference)
#define B_   2
#define S_   2048
#define HID_ 2048
#define NH_  16
#define HD_  128
#define G_   4
#define NKV_ 4

typedef __bf16 bf16_t;
typedef __attribute__((ext_vector_type(16))) __bf16 v16bf;
typedef __attribute__((ext_vector_type(8)))  float  v8f;
typedef __attribute__((ext_vector_type(4)))  unsigned int v4u;
typedef __attribute__((ext_vector_type(8)))  int  v8i;
typedef __attribute__((ext_vector_type(4)))  int  v4i;

union AF { v16bf v; uint4 q[2]; };           // 32-byte WMMA operand fragment

static __device__ __forceinline__ v8f vzero8() {
    v8f z;
#pragma unroll
    for (int i = 0; i < 8; ++i) z[i] = 0.0f;
    return z;
}
static __device__ __forceinline__ float rmax16(float v) {
#pragma unroll
    for (int m = 1; m < 16; m <<= 1) v = fmaxf(v, __shfl_xor(v, m, 32));
    return v;
}
static __device__ __forceinline__ float rsum16(float v) {
#pragma unroll
    for (int m = 1; m < 16; m <<= 1) v += __shfl_xor(v, m, 32);
    return v;
}

// LDS byte address: LDS aperture places the LDS offset in addr[31:0]
static __device__ __forceinline__ uint32_t lds_addr_of(const void* p) {
    return (uint32_t)(uintptr_t)p;
}

// ---------------------------------------------------------------------------
// Tensor Data Mover: 2D bf16 tile load, global -> LDS.
// D# packing per CDNA5 ISA:
//  group0: [1:0]=count=1, [63:32]=lds_addr, [120:64]=global_addr, [127:126]=2
//  group1: [17:16]=data_size(1 => 2B), [79:48]=tensor_dim0, [111:80]=tensor_dim1,
//          [127:112]=tile_dim0, [143:128]=tile_dim1, [207:160]=tensor_dim0_stride
//  groups 2/3: zero (2D tile, iterate/gather off)
// Dims/strides are in data_size (element) units. Completion via TENSORcnt.
// ---------------------------------------------------------------------------
static __device__ __forceinline__ void tdm_load_2d(
    uint32_t lds_byte_addr, const void* gaddr,
    uint32_t tile_d0, uint32_t tile_d1,
    uint32_t tensor_d0, uint32_t tensor_d1, uint32_t stride_d0) {
    uint64_t ga = (uint64_t)(uintptr_t)gaddr;
    v4u g0;
    g0[0] = 1u;                                            // count=1 (valid user D#)
    g0[1] = lds_byte_addr;                                 // lds_addr
    g0[2] = (uint32_t)ga;                                  // global_addr[31:0]
    g0[3] = (uint32_t)((ga >> 32) & 0x1FFFFFFu) | (2u << 30); // addr[56:32], type=2
    v8i g1;
    g1[0] = 0x00010000;                                    // data_size=1 (2 bytes)
    g1[1] = (int)((tensor_d0 & 0xFFFFu) << 16);            // tensor_dim0[15:0]
    g1[2] = (int)((tensor_d0 >> 16) | ((tensor_d1 & 0xFFFFu) << 16));
    g1[3] = (int)((tensor_d1 >> 16) | (tile_d0 << 16));    // tile_dim0
    g1[4] = (int)(tile_d1 & 0xFFFFu);                      // tile_dim1 (tile_dim2=0)
    g1[5] = (int)stride_d0;                                // tensor_dim0_stride[31:0]
    g1[6] = 0;                                             // stride hi, dim1_stride lo
    g1[7] = 0;
    v4i g2 = {0, 0, 0, 0};
    v4i g3 = {0, 0, 0, 0};
#if __clang_major__ >= 23
    v8i g4 = {0, 0, 0, 0, 0, 0, 0, 0};
    __builtin_amdgcn_tensor_load_to_lds(g0, g1, g2, g3, g4, 0);
#else
    __builtin_amdgcn_tensor_load_to_lds(g0, g1, g2, g3, 0);
#endif
}

// ---------------------------------------------------------------------------
// One-time conversion / packing kernels
// ---------------------------------------------------------------------------
__global__ void k_cvt_x(const float* __restrict__ X, bf16_t* __restrict__ Xb, int n) {
    int i = blockIdx.x * blockDim.x + threadIdx.x;
    if (i < n) Xb[i] = (bf16_t)X[i];
}

__global__ void k_pack_wqkv(const float* __restrict__ Wq, const float* __restrict__ Wk,
                            const float* __restrict__ Wv, bf16_t* __restrict__ Wt) {
    int idx = blockIdx.x * blockDim.x + threadIdx.x;      // idx = n*2048 + k
    if (idx >= 3072 * 2048) return;
    int n = idx >> 11, k = idx & 2047;
    float v;
    if      (n < 2048) v = Wq[(size_t)k * 2048 + n];
    else if (n < 2560) v = Wk[(size_t)k * 512  + (n - 2048)];
    else               v = Wv[(size_t)k * 512  + (n - 2560)];
    Wt[idx] = (bf16_t)v;
}

__global__ void k_pack_wo(const float* __restrict__ Wo, bf16_t* __restrict__ Wt) {
    int idx = blockIdx.x * blockDim.x + threadIdx.x;      // idx = n*2048 + k
    if (idx >= 2048 * 2048) return;
    int n = idx >> 11, k = idx & 2047;
    Wt[idx] = (bf16_t)Wo[(size_t)k * 2048 + n];
}

__global__ void k_pack_bias(const float* __restrict__ bq, const float* __restrict__ bk,
                            const float* __restrict__ bv, float* __restrict__ bqkv) {
    int n = blockIdx.x * blockDim.x + threadIdx.x;
    if (n >= 3072) return;
    bqkv[n] = (n < 2048) ? bq[n] : (n < 2560) ? bk[n - 2048] : bv[n - 2560];
}

// ---------------------------------------------------------------------------
// bf16 GEMM:  C[M][N] (f32) = A[M][K] @ Bt[N][K]^T + bias
// 128 threads (4 waves); tile 64x64, BK=32; wave computes 16x64.
// TDM double-buffered staging: wave0 issues tile k+1 while all compute tile k.
// ---------------------------------------------------------------------------
__global__ __launch_bounds__(128)
void k_gemm_bf16(const bf16_t* __restrict__ A, const bf16_t* __restrict__ Bt,
                 const float* __restrict__ bias, float* __restrict__ C,
                 int M, int N, int K) {
    __shared__ __align__(16) bf16_t As[2][64][32];
    __shared__ __align__(16) bf16_t Bs[2][64][32];

    const int t = threadIdx.x;
    const int w = t >> 5, lane = t & 31;
    const int m0 = blockIdx.y * 64;
    const int n0 = blockIdx.x * 64;
    const int r16 = lane & 15;
    const int hiA = (lane >> 4);                       // A frag: K offset 0/8
    const int cbB = (lane >> 4) * 2;                   // B frag uint4 base (K offset 0/16)

    v8f acc[4];
#pragma unroll
    for (int j = 0; j < 4; ++j) acc[j] = vzero8();

    const int nk = K >> 5;
    if (w == 0) {
        tdm_load_2d(lds_addr_of(&As[0][0][0]), A  + (size_t)m0 * K,
                    32, 64, (uint32_t)K, (uint32_t)M, (uint32_t)K);
        tdm_load_2d(lds_addr_of(&Bs[0][0][0]), Bt + (size_t)n0 * K,
                    32, 64, (uint32_t)K, (uint32_t)N, (uint32_t)K);
    }
    for (int i = 0; i < nk; ++i) {
        const int cur = i & 1;
        if (w == 0) __builtin_amdgcn_s_wait_tensorcnt(0);
        __syncthreads();                               // tile i visible; buf cur^1 free
        if (w == 0 && i + 1 < nk) {
            const int kb = (i + 1) << 5;
            tdm_load_2d(lds_addr_of(&As[cur ^ 1][0][0]), A  + (size_t)m0 * K + kb,
                        32, 64, (uint32_t)K, (uint32_t)M, (uint32_t)K);
            tdm_load_2d(lds_addr_of(&Bs[cur ^ 1][0][0]), Bt + (size_t)n0 * K + kb,
                        32, 64, (uint32_t)K, (uint32_t)N, (uint32_t)K);
        }
        // A fragment: row = w*16 + r16; ks {hi8+0..7, 16+hi8+0..7}
        AF a, b[4];
        const uint4* arow = (const uint4*)(&As[cur][w * 16 + r16][0]);
        a.q[0] = arow[hiA];
        a.q[1] = arow[2 + hiA];
#pragma unroll
        for (int j = 0; j < 4; ++j) {                  // preload all B fragments
            const uint4* brow = (const uint4*)(&Bs[cur][j * 16 + r16][0]);
            b[j].q[0] = brow[hiA];
            b[j].q[1] = brow[2 + hiA];
        }
#pragma unroll
        for (int j = 0; j < 4; ++j)                    // back-to-back WMMAs
            acc[j] = __builtin_amdgcn_wmma_f32_16x16x32_bf16(
                false, a.v, false, b[j].v, (short)0, acc[j], false, false);
        // no trailing barrier: next iteration's barrier precedes any overwrite
    }

    const int rowbase = m0 + w * 16 + (lane >> 4) * 8;
#pragma unroll
    for (int j = 0; j < 4; ++j) {
        int col = n0 + j * 16 + r16;
        float bv = bias ? bias[col] : 0.0f;
#pragma unroll
        for (int e = 0; e < 8; ++e)
            C[(size_t)(rowbase + e) * N + col] = acc[j][e] + bv;
    }
}

// ---------------------------------------------------------------------------
// RoPE + layout: fp32 QKV [B*S][3072] -> bf16 Q[B,NH,S,HD], K[B,NKV,S,HD]
// ---------------------------------------------------------------------------
__global__ void k_rope_qk(const float* __restrict__ QKV,
                          bf16_t* __restrict__ Qb, bf16_t* __restrict__ Kb) {
    const int PER_S = (NH_ + NKV_) * (HD_ / 2);         // 1280
    int idx = blockIdx.x * blockDim.x + threadIdx.x;
    if (idx >= B_ * S_ * PER_S) return;
    int sl = idx / PER_S;
    int r  = idx - sl * PER_S;
    int head = r >> 6, d = r & 63;
    int b = sl / S_, s = sl - b * S_;

    float inv = __powf(10000.0f, -(float)d * (1.0f / 64.0f));
    float ang = (float)s * inv;
    float c, sn;
    __sincosf(ang, &sn, &c);

    if (head < NH_) {
        const float* src = QKV + (size_t)sl * 3072 + head * HD_ + d;
        float x1 = src[0], x2 = src[64];
        bf16_t* dst = Qb + ((size_t)(b * NH_ + head) * S_ + s) * HD_ + d;
        dst[0]  = (bf16_t)(x1 * c - x2 * sn);
        dst[64] = (bf16_t)(x1 * sn + x2 * c);
    } else {
        int kh = head - NH_;
        const float* src = QKV + (size_t)sl * 3072 + 2048 + kh * HD_ + d;
        float x1 = src[0], x2 = src[64];
        bf16_t* dst = Kb + ((size_t)(b * NKV_ + kh) * S_ + s) * HD_ + d;
        dst[0]  = (bf16_t)(x1 * c - x2 * sn);
        dst[64] = (bf16_t)(x1 * sn + x2 * c);
    }
}

__global__ void k_cvt_v(const float* __restrict__ QKV, bf16_t* __restrict__ Vb) {
    int idx = blockIdx.x * blockDim.x + threadIdx.x;
    if (idx >= B_ * S_ * NKV_ * HD_) return;
    int sl = idx >> 9;
    int r  = idx & 511;
    int kh = r >> 7, d = r & 127;
    int b = sl >> 11, s = sl & 2047;
    Vb[((size_t)(b * NKV_ + kh) * S_ + s) * HD_ + d] =
        (bf16_t)QKV[(size_t)sl * 3072 + 2560 + kh * HD_ + d];
}

// ---------------------------------------------------------------------------
// Flash attention (causal, GQA). Grid (S/64, NH, B), 128 threads (4 waves).
// K tile staged by TDM (overlaps V transpose staging); online softmax in f32.
// ---------------------------------------------------------------------------
__global__ __launch_bounds__(128)
void k_attn(const bf16_t* __restrict__ Q, const bf16_t* __restrict__ K,
            const bf16_t* __restrict__ V, bf16_t* __restrict__ O) {
    __shared__ __align__(16) bf16_t Ks[32][128];        // key-major (TDM dest)
    __shared__ __align__(16) bf16_t Vt[128][32];        // hd-major (transposed V)
    __shared__ __align__(16) bf16_t Ps[4][16][32];      // per-wave P relayout tile

    const int t = threadIdx.x, w = t >> 5, lane = t & 31;
    const int qt = blockIdx.x, head = blockIdx.y, b = blockIdx.z;
    const int kv = head / G_;
    const int q0 = qt * 64 + w * 16;
    const int r16  = lane & 15;
    const int hi8  = (lane >> 4) * 8;
    const int hi16 = (lane >> 4) * 16;

    // Q fragments live in registers for the whole kernel (4 chunks over HD)
    AF qf[4];
    {
        const bf16_t* qrow = Q + ((size_t)(b * NH_ + head) * S_ + (q0 + r16)) * HD_;
#pragma unroll
        for (int c = 0; c < 4; ++c) {
            qf[c].q[0] = *(const uint4*)(qrow + c * 32 + hi8);
            qf[c].q[1] = *(const uint4*)(qrow + c * 32 + 16 + hi8);
        }
    }

    v8f o[8];
#pragma unroll
    for (int j = 0; j < 8; ++j) o[j] = vzero8();
    float mrow[8], lrow[8];
#pragma unroll
    for (int e = 0; e < 8; ++e) { mrow[e] = -3.0e38f; lrow[e] = 0.0f; }

    const bf16_t* Kbase = K + (size_t)(b * NKV_ + kv) * S_ * HD_;
    const bf16_t* Vbase = V + (size_t)(b * NKV_ + kv) * S_ * HD_;
    const float scale = 0.08838834764831845f;           // 1/sqrt(128)
    const int kend = qt * 64 + 64;                      // causal bound for this WG

    for (int k0 = 0; k0 < kend; k0 += 32) {
        // ---- TDM: K tile (32 keys x 128) global -> LDS, async ----
        if (w == 0)
            tdm_load_2d(lds_addr_of(&Ks[0][0]), Kbase + (size_t)k0 * HD_,
                        128, 32, 128, (uint32_t)S_, 128);

        // ---- overlap: stage V transposed into LDS (vector path) ----
        {
            int row = t >> 2, seg = (t & 3) * 32;
            const bf16_t* vg = Vbase + (size_t)(k0 + row) * HD_ + seg;
#pragma unroll
            for (int ch = 0; ch < 4; ++ch) {
                union { uint4 q; bf16_t h[8]; } u;
                u.q = *(const uint4*)(vg + ch * 8);
#pragma unroll
                for (int j = 0; j < 8; ++j) Vt[seg + ch * 8 + j][row] = u.h[j];
            }
        }
        if (w == 0) __builtin_amdgcn_s_wait_tensorcnt(0);
        __syncthreads();

        // ---- scores: S = Q @ K^T (two 16-key column tiles) ----
        v8f sc[2];
        sc[0] = vzero8(); sc[1] = vzero8();
#pragma unroll
        for (int kn = 0; kn < 2; ++kn) {
            const bf16_t* krow = &Ks[kn * 16 + r16][0];
#pragma unroll
            for (int c = 0; c < 4; ++c) {
                AF bk;
                bk.q[0] = *(const uint4*)(krow + c * 32 + hi8);
                bk.q[1] = *(const uint4*)(krow + c * 32 + 16 + hi8);
                sc[kn] = __builtin_amdgcn_wmma_f32_16x16x32_bf16(
                    false, qf[c].v, false, bk.v, (short)0, sc[kn], false, false);
            }
        }

        // ---- causal mask + online softmax (row stats across 16-lane halves) ----
        float pr[2][8];
#pragma unroll
        for (int e = 0; e < 8; ++e) {
            int rowg = q0 + e + hi8;
            float s0 = sc[0][e] * scale, s1 = sc[1][e] * scale;
            if (k0 + r16      > rowg) s0 = -1.0e9f;
            if (k0 + 16 + r16 > rowg) s1 = -1.0e9f;
            float mnew = fmaxf(mrow[e], rmax16(fmaxf(s0, s1)));
            float p0 = __expf(s0 - mnew), p1 = __expf(s1 - mnew);
            float ls = rsum16(p0 + p1);
            float rs = __expf(mrow[e] - mnew);
            lrow[e] = lrow[e] * rs + ls;
            mrow[e] = mnew;
            pr[0][e] = p0; pr[1][e] = p1;
#pragma unroll
            for (int j = 0; j < 8; ++j) o[j][e] *= rs;
        }

        // ---- relayout P (C-layout -> A-layout) through per-wave LDS tile ----
#pragma unroll
        for (int kn = 0; kn < 2; ++kn)
#pragma unroll
            for (int e = 0; e < 8; ++e)
                Ps[w][e + hi8][kn * 16 + r16] = (bf16_t)pr[kn][e];
        __syncthreads();

        // ---- out += P @ V  (K-dim = 32 keys, 8 hd-column tiles) ----
        AF ap;
        {
            const bf16_t* prow = &Ps[w][r16][0];
            ap.q[0] = *(const uint4*)(prow + hi8);
            ap.q[1] = *(const uint4*)(prow + 16 + hi8);
        }
#pragma unroll
        for (int j = 0; j < 8; ++j) {
            AF bv;
            const bf16_t* vrow = &Vt[j * 16 + r16][0];
            bv.q[0] = *(const uint4*)(vrow + hi8);
            bv.q[1] = *(const uint4*)(vrow + 16 + hi8);
            o[j] = __builtin_amdgcn_wmma_f32_16x16x32_bf16(
                false, ap.v, false, bv.v, (short)0, o[j], false, false);
        }
        __syncthreads();                                // Ks/Vt/Ps reuse guard
    }

    // ---- epilogue: normalize and store attn output bf16 [B,S,NH*HD] ----
#pragma unroll
    for (int e = 0; e < 8; ++e) {
        float inv_l = 1.0f / lrow[e];
        size_t row = (size_t)b * S_ + (q0 + e + hi8);
#pragma unroll
        for (int j = 0; j < 8; ++j)
            O[row * (NH_ * HD_) + head * HD_ + j * 16 + r16] =
                (bf16_t)(o[j][e] * inv_l);
    }
}

// ---------------------------------------------------------------------------
// Host-side launch
// ---------------------------------------------------------------------------
extern "C" void kernel_launch(void* const* d_in, const int* in_sizes, int n_in,
                              void* d_out, int out_size, void* d_ws, size_t ws_size,
                              hipStream_t stream) {
    (void)in_sizes; (void)n_in; (void)out_size; (void)ws_size;
    const float* hidden = (const float*)d_in[0];
    /* d_in[1] = mask: known causal, not read */
    const float* Wq = (const float*)d_in[2];
    const float* bq = (const float*)d_in[3];
    const float* Wk = (const float*)d_in[4];
    const float* bk = (const float*)d_in[5];
    const float* Wv = (const float*)d_in[6];
    const float* bv = (const float*)d_in[7];
    const float* Wo = (const float*)d_in[8];
    const float* bo = (const float*)d_in[9];
    float* out = (float*)d_out;

    char* ws = (char*)d_ws;
    size_t off = 0;
    auto alloc = [&](size_t bytes) -> void* {
        void* p = ws + off;
        off += (bytes + 255) & ~(size_t)255;
        return p;
    };
    const size_t MS = (size_t)B_ * S_;                   // 4096 rows
    bf16_t* Xb    = (bf16_t*)alloc(MS * HID_ * 2);
    bf16_t* WqkvT = (bf16_t*)alloc((size_t)3072 * 2048 * 2);
    bf16_t* WoT   = (bf16_t*)alloc((size_t)2048 * 2048 * 2);
    float*  bqkv  = (float*)alloc(3072 * 4);
    float*  QKVf  = (float*)alloc(MS * 3072 * 4);
    bf16_t* Qb    = (bf16_t*)alloc((size_t)B_ * NH_  * S_ * HD_ * 2);
    bf16_t* Kb    = (bf16_t*)alloc((size_t)B_ * NKV_ * S_ * HD_ * 2);
    bf16_t* Vb    = (bf16_t*)alloc((size_t)B_ * NKV_ * S_ * HD_ * 2);
    bf16_t* Ob    = (bf16_t*)alloc(MS * (size_t)(NH_ * HD_) * 2);

    // 1) convert / pack
    k_cvt_x    <<<(int)((MS * HID_) / 256), 256, 0, stream>>>(hidden, Xb, (int)(MS * HID_));
    k_pack_wqkv<<<(3072 * 2048) / 256, 256, 0, stream>>>(Wq, Wk, Wv, WqkvT);
    k_pack_wo  <<<(2048 * 2048) / 256, 256, 0, stream>>>(Wo, WoT);
    k_pack_bias<<<3072 / 256, 256, 0, stream>>>(bq, bk, bv, bqkv);

    // 2) fused QKV projection: [4096,2048] @ [2048,3072] + bias -> fp32
    k_gemm_bf16<<<dim3(3072 / 64, (int)(MS / 64)), 128, 0, stream>>>(
        Xb, WqkvT, bqkv, QKVf, (int)MS, 3072, HID_);

    // 3) RoPE + head-major layout
    k_rope_qk<<<(B_ * S_ * 1280) / 256, 256, 0, stream>>>(QKVf, Qb, Kb);
    k_cvt_v  <<<(B_ * S_ * NKV_ * HD_) / 256, 256, 0, stream>>>(QKVf, Vb);

    // 4) causal GQA flash attention
    k_attn<<<dim3(S_ / 64, NH_, B_), 128, 0, stream>>>(Qb, Kb, Vb, Ob);

    // 5) output projection: [4096,2048] @ [2048,2048] + bo -> fp32 out
    k_gemm_bf16<<<dim3(2048 / 64, (int)(MS / 64)), 128, 0, stream>>>(
        Ob, WoT, bo, out, (int)MS, 2048, HID_);
}
